// DeepBSDESC_25675314495642
// MI455X (gfx1250) — matrix operand order repaired and checked by reading, not compiled, not run on hardware
//
#include <hip/hip_runtime.h>
#include <hip/hip_bf16.h>
#include <math.h>

// ---------------- problem constants (match reference) ----------------
#define NSTEPS 128
#define BPATHS 32768
#define HDIM   64
#define DTSTEP (1.0f / 128.0f)

typedef __attribute__((ext_vector_type(16))) _Float16 v16h;
typedef __attribute__((ext_vector_type(8)))  float    v8f;

#define WAVES_PER_BLOCK 4
#define PATHS_PER_WAVE  16
#define PATHS_PER_BLOCK (WAVES_PER_BLOCK * PATHS_PER_WAVE)
#define NT_THREADS (WAVES_PER_BLOCK * 32)

// Branchless tanh on the native transcendental pipe:
// tanh(x) = 1 - 2/(e^{2x}+1);  e^{2x} = exp2(x * 2*log2(e)).
// Saturates correctly: x>>0 -> exp2=inf -> rcp=0 -> 1; x<<0 -> rcp(1)=1 -> -1.
__device__ __forceinline__ float fast_tanh(float x) {
  const float e = __builtin_amdgcn_exp2f(x * 2.885390081777927f);
  return 1.0f - 2.0f * __builtin_amdgcn_rcpf(e + 1.0f);
}

// Build WMMA A-fragment (16 paths x K=32, fp16) holding the 5 features,
// zero-padded. A layout: lane<16 -> M=lane, halfs 0..7 = K0..7 (features),
// halfs 8..15 = K16..23 (zero); lanes>=16 cover K8..15/24..31 (all zero).
__device__ __forceinline__ v16h make_feat(float f0, float f1, float f2,
                                          float f3, float f4, int lane) {
  v16h a;
#pragma unroll
  for (int e = 0; e < 16; ++e) a[e] = (_Float16)0.0f;
  if (lane < 16) {
    a[0] = (_Float16)f0;
    a[1] = (_Float16)f1;
    a[2] = (_Float16)f2;
    a[3] = (_Float16)f3;
    a[4] = (_Float16)f4;
  }
  return a;
}

// One 16-path MLP: o = tanh(ft @ W1 + b1) @ W2 + b2.
// B1: LDS fp16 fragments [4 tiles][32 lanes][16 halfs] (pre-padded).
// B2: LDS fp16 fragments [2 ktiles][32 lanes][16 halfs] (pre-padded).
// hA: per-wave LDS fp16 staging in A-fragment layout [2][32][16].
// sOut: per-wave LDS f32 [16 paths][8].
// All WMMA calls sit in uniform control flow (EXEC all ones).
template <int NOUT>
__device__ __forceinline__ void mlp16(v16h afrag,
                                      const _Float16* __restrict__ B1,
                                      const float* __restrict__ sB1,
                                      const _Float16* __restrict__ B2,
                                      const float* __restrict__ sB2,
                                      _Float16* __restrict__ hA,
                                      float* __restrict__ sOut,
                                      int lane, float* out) {
  const int m  = lane & 15;
  const int hi = lane >> 4;

  // ---------- layer 1: 4 N-tiles of 16, K=32 (features padded) ----------
  v8f hacc[4];
#pragma unroll
  for (int nt = 0; nt < 4; ++nt) {
    const v16h b = *(const v16h*)(B1 + ((nt * 32 + lane) << 4));
    v8f c = {};
    c = __builtin_amdgcn_wmma_f32_16x16x32_f16(false, afrag, false, b,
                                               (short)0, c, false, false);
    hacc[nt] = c;
  }

  // bias + tanh; store directly into the layer-2 A-fragment fp16 layout.
  // Produced value sits at (M = r+8*hi, Khidden = nt*16 + m); its A slot is
  // kt = Khidden/32, lane' = M + 16*((Khidden%32)/16),
  // e = Khidden%16.
#pragma unroll
  for (int nt = 0; nt < 4; ++nt) {
    const int Kv  = nt * 16 + m;
    const int kt  = nt >> 1;
    const int k32 = Kv & 31;
    const int e   = k32 & 15;
    const int lh  = k32 >> 4;
    const float bias = sB1[Kv];
    _Float16* dst = hA + ((kt * 32 + 16 * lh) << 4) + e;
#pragma unroll
    for (int r = 0; r < 8; ++r) {
      const int Mv = r + 8 * hi;
      dst[Mv << 4] = (_Float16)fast_tanh(hacc[nt][r] + bias);
    }
  }
  __builtin_amdgcn_wave_barrier();  // per-wave staging: same-wave LDS is in-order

  // ---------- layer 2: K=64 -> 2 chained WMMAs, N padded to 16 ----------
  v8f c2 = {};
#pragma unroll
  for (int kt = 0; kt < 2; ++kt) {
    const v16h a2 = *(const v16h*)(hA + ((kt * 32 + lane) << 4));
    const v16h b2 = *(const v16h*)(B2 + ((kt * 32 + lane) << 4));
    c2 = __builtin_amdgcn_wmma_f32_16x16x32_f16(false, a2, false, b2,
                                                (short)0, c2, false, false);
  }
  __builtin_amdgcn_wave_barrier();

  // scatter o[M][N<NOUT] (+bias) to per-wave LDS, gather per path on lanes<16
  if (m < NOUT) {
    const float bias = sB2[m];
#pragma unroll
    for (int r = 0; r < 8; ++r) {
      const int Mv = r + 8 * hi;
      sOut[Mv * 8 + m] = c2[r] + bias;
    }
  }
  __builtin_amdgcn_wave_barrier();
  if (lane < 16) {
#pragma unroll
    for (int j = 0; j < NOUT; ++j) out[j] = sOut[lane * 8 + j];
  }
  __builtin_amdgcn_wave_barrier();
}

__global__ __launch_bounds__(NT_THREADS, 1)
void bsde_scan_kernel(const float* __restrict__ pN, const float* __restrict__ pX,
                      const float* __restrict__ pp, const float* __restrict__ pt,
                      const float* __restrict__ dBg,
                      const float* __restrict__ Wr1, const float* __restrict__ br1,
                      const float* __restrict__ Wr2, const float* __restrict__ br2,
                      const float* __restrict__ Wg1, const float* __restrict__ bg1,
                      const float* __restrict__ Wg2, const float* __restrict__ bg2,
                      const float* __restrict__ Wj1, const float* __restrict__ bj1,
                      const float* __restrict__ Wj2, const float* __restrict__ bj2,
                      float* __restrict__ outp) {
  // fp16 B fragments, pre-laid-out in WMMA register order, zero-padded once.
  __shared__ __align__(32) _Float16 hBg1[4 * 32 * 16];  // grad layer1
  __shared__ __align__(32) _Float16 hBj1[4 * 32 * 16];  // jump layer1
  __shared__ __align__(32) _Float16 hBg2[2 * 32 * 16];  // grad layer2 (N pad 16)
  __shared__ __align__(32) _Float16 hBj2[2 * 32 * 16];  // jump layer2
  __shared__ float sbg1[HDIM], sbg2[8], sbj1[HDIM], sbj2[8];
  // per-wave staging
  __shared__ __align__(32) _Float16 hA[WAVES_PER_BLOCK][2 * 32 * 16];
  __shared__ float sOut[WAVES_PER_BLOCK][16 * 8];

  const int tid  = threadIdx.x;
  const int lane = tid & 31;
  const int wave = tid >> 5;
  _Float16* myA  = hA[wave];
  float*    myO  = sOut[wave];
  const int pbase = blockIdx.x * PATHS_PER_BLOCK + wave * PATHS_PER_WAVE;
  const int path  = pbase + (lane & 15);

  (void)pt;  // discrete_t == linspace(0,1,129) -> t_k = k*DT exactly

  // ---- one-time zero fill of fragment padding (never rewritten) ----
  for (int i = tid; i < 4 * 32 * 16; i += NT_THREADS) {
    hBg1[i] = (_Float16)0.0f;
    hBj1[i] = (_Float16)0.0f;
  }
  for (int i = tid; i < 2 * 32 * 16; i += NT_THREADS) {
    hBg2[i] = (_Float16)0.0f;
    hBj2[i] = (_Float16)0.0f;
  }
  // ---- stage u0 weights (Wr) into the grad slots ----
  for (int i = tid; i < 5 * HDIM; i += NT_THREADS) {
    const int K = i >> 6, N = i & 63;           // W1[k][n], k<5 -> lanes<16
    hBg1[(((N >> 4) * 32 + (N & 15)) << 4) + K] = (_Float16)Wr1[i];
  }
  for (int i = tid; i < HDIM * 2; i += NT_THREADS) {
    const int K = i >> 1, N = i & 1;            // W2[k][n], NOUT=2
    const int kt = K >> 5, k32 = K & 31;
    hBg2[((kt * 32 + (N + 16 * (k32 >> 4))) << 4) + (k32 & 15)] = (_Float16)Wr2[i];
  }
  for (int i = tid; i < HDIM; i += NT_THREADS) sbg1[i] = br1[i];
  if (tid < 2) sbg2[tid] = br2[tid];
  __syncthreads();

  // ---------------- u0 = mlp_c(Wr..)(feat at step 0) ----------------
  float n0 = 0.f, xa = 0.f, xb = 0.f, xc = 0.f, p0 = 0.f;
  if (lane < 16) {
    n0 = pN[path];
    xa = pX[(size_t)path * 3 + 0];
    xb = pX[(size_t)path * 3 + 1];
    xc = pX[(size_t)path * 3 + 2];
    p0 = pp[path];
  }
  v16h a0 = make_feat(n0, xa, xb, xc, p0, lane);
  float u0v[2] = {0.f, 0.f};
  mlp16<2>(a0, hBg1, sbg1, hBg2, sbg2, myA, myO, lane, u0v);

  float ur = u0v[0], uim = u0v[1];  // complex carry u
  float acc = 0.f;                  // phase accumulator: sum p_s * |x_s|^2

  // ---------------- sequential scan over 128 steps ----------------
#pragma unroll 1
  for (int t = 0; t < NSTEPS; ++t) {
    __syncthreads();  // all waves finished reading previous step's weights
    {
      const float* wg1 = Wg1 + (size_t)t * 5 * HDIM;
      const float* wg2 = Wg2 + (size_t)t * HDIM * 6;
      const float* wj1 = Wj1 + (size_t)t * 5 * HDIM;
      const float* wj2 = Wj2 + (size_t)t * HDIM * 2;
      // layer-1 weights: slot(K<5, N): tile=N>>4, lane=N&15, e=K
      for (int i = tid; i < 5 * HDIM; i += NT_THREADS) {
        const int K = i >> 6, N = i & 63;
        const int s = (((N >> 4) * 32 + (N & 15)) << 4) + K;
        hBg1[s] = (_Float16)wg1[i];
        hBj1[s] = (_Float16)wj1[i];
      }
      // layer-2 weights: slot(K, N): kt=K>>5, lane=N+16*((K&31)>>4), e=K&15
      for (int i = tid; i < HDIM * 6; i += NT_THREADS) {
        const int K = i / 6, N = i % 6;
        const int k32 = K & 31;
        hBg2[(((K >> 5) * 32 + (N + 16 * (k32 >> 4))) << 4) + (k32 & 15)] =
            (_Float16)wg2[i];
      }
      for (int i = tid; i < HDIM * 2; i += NT_THREADS) {
        const int K = i >> 1, N = i & 1;
        const int k32 = K & 31;
        hBj2[(((K >> 5) * 32 + (N + 16 * (k32 >> 4))) << 4) + (k32 & 15)] =
            (_Float16)wj2[i];
      }
      for (int i = tid; i < HDIM; i += NT_THREADS) {
        sbg1[i] = bg1[(size_t)t * HDIM + i];
        sbj1[i] = bj1[(size_t)t * HDIM + i];
      }
      if (tid < 6) sbg2[tid] = bg2[(size_t)t * 6 + tid];
      if (tid < 2) sbj2[tid] = bj2[(size_t)t * 2 + tid];
      if (lane == 0 && t + 1 < NSTEPS) {  // warm caches for next step's weights
        __builtin_prefetch(Wg1 + (size_t)(t + 1) * 5 * HDIM, 0, 1);
        __builtin_prefetch(Wg2 + (size_t)(t + 1) * HDIM * 6, 0, 1);
        __builtin_prefetch(Wj1 + (size_t)(t + 1) * 5 * HDIM, 0, 1);
        __builtin_prefetch(Wj2 + (size_t)(t + 1) * HDIM * 2, 0, 1);
      }
    }
    __syncthreads();

    // per-path state for this step (lane p<16 owns path pbase+p)
    float n_ = 0.f, x0 = 0.f, x1 = 0.f, x2 = 0.f, p_ = 0.f;
    float dB0 = 0.f, dB1 = 0.f, dB2 = 0.f, nNext = 0.f;
    if (lane < 16) {
      const size_t off = (size_t)t * BPATHS + path;
      n_ = pN[off];
      nNext = pN[off + BPATHS];
      p_ = pp[off];
      x0 = pX[off * 3 + 0];
      x1 = pX[off * 3 + 1];
      x2 = pX[off * 3 + 2];
      dB0 = dBg[off * 3 + 0];
      dB1 = dBg[off * 3 + 1];
      dB2 = dBg[off * 3 + 2];
    }

    v16h aft = make_feat(n_, x0, x1, x2, p_, lane);
    v16h aftp = aft, aftm = aft;
    if (lane < 16) {
      aftp[0] = (_Float16)(n_ + 1.0f);
      aftm[0] = (_Float16)(n_ - 1.0f);
    }

    float og[6] = {0, 0, 0, 0, 0, 0};
    float oj[2] = {0, 0}, ojp[2] = {0, 0}, ojm[2] = {0, 0};
    mlp16<6>(aft,  hBg1, sbg1, hBg2, sbg2, myA, myO, lane, og);   // grad_u
    mlp16<2>(aft,  hBj1, sbj1, hBj2, sbj2, myA, myO, lane, oj);   // u_i
    mlp16<2>(aftp, hBj1, sbj1, hBj2, sbj2, myA, myO, lane, ojp);  // u_{i+1}
    mlp16<2>(aftm, hBj1, sbj1, hBj2, sbj2, myA, myO, lane, ojm);  // u_{i-1}

    if (lane < 16) {
      // ef_t = exp(-i * DT * acc)  (functional is purely imaginary)
      const float theta = DTSTEP * acc;
      const float ct = __cosf(theta), st = __sinf(theta);
      const float efr = ct, efi = -st;

      // grad_bmm = k * grad_u . (Ti * S * dB)   (M = Ti @ S^T, S symmetric)
      const float kk = __builtin_amdgcn_sqrtf(2.0f * (0.5f + 0.1f * fabsf(n_)));
      const float xdB = x0 * dB0 + x1 * dB1 + x2 * dB2;
      const float w0 = 0.5f * dB0 + 0.1f * x0 * xdB;  // S @ dB
      const float w1 = 0.5f * dB1 + 0.1f * x1 * xdB;
      const float w2 = 0.5f * dB2 + 0.1f * x2 * xdB;
      const float nrm = 1.0f + x0 * x0 + x1 * x1 + x2 * x2;
      const float xw = (x0 * w0 + x1 * w1 + x2 * w2) * __builtin_amdgcn_rcpf(nrm);
      const float v0 = w0 - x0 * xw;                  // Ti @ (S @ dB)
      const float v1 = w1 - x1 * xw;
      const float v2 = w2 - x2 * xw;
      const float gbr = kk * (og[0] * v0 + og[1] * v1 + og[2] * v2);
      const float gbi = kk * (og[3] * v0 + og[4] * v1 + og[5] * v2);

      // jump differences
      const float dupr = ojp[0] - oj[0], dupi = ojp[1] - oj[1];
      const float dumr = ojm[0] - oj[0], dumi = ojm[1] - oj[1];

      // fv = p*(u/ef) + 0.1*(1+t)*sum(grad_u);  u/ef = u*exp(+i*theta)
      const float yr = ur * ct - uim * st;
      const float yi = ur * st + uim * ct;
      const float tcur = (float)t * DTSTEP;
      const float fvr = p_ * yr + 0.1f * (1.0f + tcur) * (og[0] + og[1] + og[2]);
      const float fvi = p_ * yi + 0.1f * (1.0f + tcur) * (og[3] + og[4] + og[5]);

      const float al = 0.5f * (n_ + 1.0f);
      const float be = 0.4f * fabsf(n_) + 0.1f;
      float dr = gbr - (al * dupr + be * dumr) * DTSTEP - fvr * DTSTEP;
      float di = gbi - (al * dupi + be * dumi) * DTSTEP - fvi * DTSTEP;

      const float dN = rintf(nNext - n_);
      if (dN > 0.0f) { dr = dupr; di = dupi; }
      else if (dN < 0.0f) { dr = dumr; di = dumi; }

      // u += ef * delt
      ur  += efr * dr - efi * di;
      uim += efr * di + efi * dr;

      // advance phase: functional_{t+1} uses p_t * U(x_t)
      acc += p_ * (x0 * x0 + x1 * x1 + x2 * x2);
    }
  }

  // ---------------- terminal payoff + output ----------------
  if (lane < 16) {
    const size_t offT = (size_t)NSTEPS * BPATHS + path;
    const float xs = pX[offT * 3 + 0] + pX[offT * 3 + 1] + pX[offT * 3 + 2];
    const float thf = DTSTEP * acc;  // exp_functional[-1] = exp(-i*thf)
    outp[2 * (size_t)path + 0] = ur;
    outp[2 * (size_t)path + 1] = uim;
    outp[2 * (size_t)BPATHS + 2 * (size_t)path + 0] =  xs * __cosf(thf);
    outp[2 * (size_t)BPATHS + 2 * (size_t)path + 1] = -xs * __sinf(thf);
  }
}

extern "C" void kernel_launch(void* const* d_in, const int* in_sizes, int n_in,
                              void* d_out, int out_size, void* d_ws, size_t ws_size,
                              hipStream_t stream) {
  (void)in_sizes; (void)n_in; (void)d_ws; (void)ws_size; (void)out_size;
  const float* pN  = (const float*)d_in[0];   // process_N  (129,B,1)
  const float* pX  = (const float*)d_in[1];   // process_X  (129,B,3)
  const float* pp  = (const float*)d_in[2];   // discrete_p (129,B,1)
  const float* pt  = (const float*)d_in[3];   // discrete_t (129,)
  const float* dB  = (const float*)d_in[4];   // delta_B    (128,B,3)
  const float* Wr1 = (const float*)d_in[5];
  const float* br1 = (const float*)d_in[6];
  const float* Wr2 = (const float*)d_in[7];
  const float* br2 = (const float*)d_in[8];
  const float* Wg1 = (const float*)d_in[9];
  const float* bg1 = (const float*)d_in[10];
  const float* Wg2 = (const float*)d_in[11];
  const float* bg2 = (const float*)d_in[12];
  const float* Wj1 = (const float*)d_in[13];
  const float* bj1 = (const float*)d_in[14];
  const float* Wj2 = (const float*)d_in[15];
  const float* bj2 = (const float*)d_in[16];
  float* outp = (float*)d_out;

  dim3 grid(BPATHS / PATHS_PER_BLOCK);
  dim3 block(NT_THREADS);
  bsde_scan_kernel<<<grid, block, 0, stream>>>(pN, pX, pp, pt, dB,
                                               Wr1, br1, Wr2, br2,
                                               Wg1, bg1, Wg2, bg2,
                                               Wj1, bj1, Wj2, bj2, outp);
}
